// Attention_78743930405585
// MI455X (gfx1250) — matrix operand order repaired and checked by reading, not compiled, not run on hardware
//
#include <hip/hip_runtime.h>
#include <stdint.h>

// ---------------------------------------------------------------------------
// Attention (B=4, S=2048, H=1024) via bf16 WMMA pipeline on gfx1250.
// Pipeline: cvt(x,W)->bf16 ; QKV GEMM (NT) ; scores GEMM (NT, scaled) ;
//           softmax ; PV GEMM (NT, V stored transposed) -> fp32 out.
//
// GEMM: block tile 128x256, BK=32, 256 threads (8 waves), wave tile 64x64
//       (16 WMMA per k-chunk per wave). Tile staging (best-first):
//         1) TDM tensor_load_to_lds (TENSORcnt, double buffered)
//         2) per-lane async global->LDS (ASYNCcnt, double buffered)
//         3) plain global->VGPR->LDS
//
// Workspace layout (bytes):
//   xbf   @ 0          : 8192*1024*2      = 16,777,216
//   wbf   @ 16777216   : 3072*1024*2      =  6,291,456
//   qbf   @ 23068672   : 4*2048*1024*2    = 16,777,216
//   kbf   @ 39845888   : 4*2048*1024*2    = 16,777,216
//   vtbf  @ 56623104   : 4*1024*2048*2    = 16,777,216  (transposed V)
//   sbuf  @ 73400320   : 4*2048*2048*4    = 67,108,864  (fp32 scores)
//   pbf   @ 140509184  : 4*2048*2048*2    = 33,554,432  (bf16 probs)
//   total 174,063,616 bytes
// ---------------------------------------------------------------------------

typedef __attribute__((ext_vector_type(16))) __bf16 v16bf;
typedef __attribute__((ext_vector_type(8)))  float  v8f;

#if defined(__has_builtin)
#if __has_builtin(__builtin_amdgcn_tensor_load_to_lds)
#define USE_TDM 1
#endif
#if __has_builtin(__builtin_amdgcn_global_load_async_to_lds_b128)
#define USE_ASYNC_LDS 1
#endif
#endif
#ifndef USE_TDM
#define USE_TDM 0
#endif
#ifndef USE_ASYNC_LDS
#define USE_ASYNC_LDS 0
#endif

__device__ __forceinline__ uint16_t f2bf(float f) {
    union { float f; uint32_t u; } v; v.f = f;
    uint32_t r = v.u + 0x7FFFu + ((v.u >> 16) & 1u);   // round-to-nearest-even
    return (uint16_t)(r >> 16);
}

union FragU { uint4 q[2]; v16bf v; };

// A fragment: elems 0..7 = s[0..7], elems 8..15 = s[16..23] (per ISA A layout)
__device__ __forceinline__ v16bf load_frag_a(const uint16_t* s) {
    FragU u;
    u.q[0] = *(const uint4*)(s);
    u.q[1] = *(const uint4*)(s + 16);
    return u.v;
}
// B fragment: elems 0..7 = s[0..7], elems 8..15 = s[16..23] (same k order as A)
__device__ __forceinline__ v16bf load_frag_b(const uint16_t* s) {
    FragU u;
    u.q[0] = *(const uint4*)(s);
    u.q[1] = *(const uint4*)(s + 16);
    return u.v;
}

#if USE_TDM
// ---- Tensor Data Mover: 2D tile (tile_rows x 32 bf16) global -> LDS -------
typedef unsigned int u32x4 __attribute__((vector_size(16)));
typedef int i32x8 __attribute__((vector_size(32)));
typedef int i32x4 __attribute__((vector_size(16)));

__device__ __forceinline__ void tdm_load_2d(const uint16_t* gsrc, const uint16_t* ldst,
                                            uint32_t tile_rows, uint32_t stride_elems)
{
    const uint64_t ga = (uint64_t)(uintptr_t)gsrc;
    const uint32_t la = (uint32_t)(uintptr_t)ldst;
    // D# group0: [1:0]=count=1 | [63:32]=lds_addr | [120:64]=global_addr | [127:126]=type 2
    u32x4 g0 = { 1u, la, (uint32_t)ga,
                 (uint32_t)((ga >> 32) & 0x01FFFFFFu) | 0x80000000u };
    // D# group1: data_size=1(2B); tensor_dim0=32; tensor_dim1=tile_rows;
    //            tile_dim0=32; tile_dim1=tile_rows; tensor_dim0_stride=stride
    const uint32_t td0 = 32u, td1 = tile_rows, tile0 = 32u, tile1 = tile_rows;
    i32x8 g1 = {
        (int)0x00010000u,                                              // ds=2B
        (int)((td0 & 0xFFFFu) << 16),                                  // td0 lo
        (int)(((td0 >> 16) & 0xFFFFu) | ((td1 & 0xFFFFu) << 16)),      // td0 hi | td1 lo
        (int)(((td1 >> 16) & 0xFFFFu) | ((tile0 & 0xFFFFu) << 16)),    // td1 hi | tile0
        (int)(tile1 & 0xFFFFu),                                        // tile1 (tile2=0)
        (int)stride_elems,                                             // stride lo32
        0, 0
    };
    i32x4 gz4 = { 0, 0, 0, 0 };
    i32x8 gz8 = { 0, 0, 0, 0, 0, 0, 0, 0 };
    __builtin_amdgcn_tensor_load_to_lds(g0, g1, gz4, gz4, gz8, 0);
}

__device__ __forceinline__ void wait_tensorcnt_le1() {
#if defined(__has_builtin) && __has_builtin(__builtin_amdgcn_s_wait_tensorcnt)
    __builtin_amdgcn_s_wait_tensorcnt((short)1);
#else
    asm volatile("s_wait_tensorcnt 0x1" ::: "memory");
#endif
}
__device__ __forceinline__ void wait_tensorcnt_0() {
#if defined(__has_builtin) && __has_builtin(__builtin_amdgcn_s_wait_tensorcnt)
    __builtin_amdgcn_s_wait_tensorcnt((short)0);
#else
    asm volatile("s_wait_tensorcnt 0x0" ::: "memory");
#endif
}
#endif

#if USE_ASYNC_LDS
typedef int g4i __attribute__((vector_size(16)));
typedef __attribute__((address_space(1))) g4i* g4i_global_ptr;
typedef __attribute__((address_space(3))) g4i* g4i_lds_ptr;

__device__ __forceinline__ void async_b128(const uint16_t* g, uint16_t* l) {
    __builtin_amdgcn_global_load_async_to_lds_b128(
        (g4i_global_ptr)(uintptr_t)g,
        (g4i_lds_ptr)(uint32_t)(uintptr_t)l,
        0, 0);
}
#endif

// ---------------------------------------------------------------------------
// Converters
// ---------------------------------------------------------------------------
__global__ __launch_bounds__(256)
void cvt_f32_bf16(const float* __restrict__ in, uint16_t* __restrict__ out, int n) {
    int i = blockIdx.x * 256 + threadIdx.x;
    if (i < n) out[i] = f2bf(in[i]);
}

__global__ __launch_bounds__(256)
void pack_w(const float* __restrict__ wq, const float* __restrict__ wk,
            const float* __restrict__ wv, uint16_t* __restrict__ out) {
    int i = blockIdx.x * 256 + threadIdx.x;          // 0 .. 3*1048576-1
    int sel = i >> 20, r = i & 1048575;
    const float* src = (sel == 0) ? wq : (sel == 1) ? wk : wv;
    out[i] = f2bf(src[r]);
}

// ---------------------------------------------------------------------------
// NT GEMM: C[m,n] = sum_k A[m,k]*B[n,k], bf16 inputs, fp32 accum via WMMA.
// MODE 0: QKV projection epilogue (bias add, bf16 out, V transposed)
// MODE 1: scores epilogue (scale by 1/32, fp32 out, batched via blockIdx.z)
// MODE 2: PV epilogue (fp32 out, batched via blockIdx.z)
// ---------------------------------------------------------------------------
template <int MODE>
__global__ __launch_bounds__(256)
void gemm_nt(const uint16_t* __restrict__ A, const uint16_t* __restrict__ B, int K,
             const float* __restrict__ bq, const float* __restrict__ bk,
             const float* __restrict__ bv,
             uint16_t* __restrict__ oQ, uint16_t* __restrict__ oK,
             uint16_t* __restrict__ oVt, float* __restrict__ oF)
{
#if USE_TDM || USE_ASYNC_LDS
    __shared__ uint16_t As[2][128 * 32];
    __shared__ uint16_t Bs[2][256 * 32];
#else
    __shared__ uint16_t As[1][128 * 32];
    __shared__ uint16_t Bs[1][256 * 32];
#endif

    const int tid  = threadIdx.x;
    const int lane = tid & 31;
    const int w    = tid >> 5;
    const int wm   = w & 1;        // 0..1 -> 64-row M sub-tile
    const int wn   = w >> 1;       // 0..3 -> 64-col N sub-tile
    const int m0   = blockIdx.x * 128;
    const int n0   = blockIdx.y * 256;
    const int bat  = blockIdx.z;

    size_t aBase = 0, bBase = 0;
    if (MODE == 1) { aBase = (size_t)bat * 2048 * 1024; bBase = (size_t)bat * 2048 * 1024; }
    if (MODE == 2) { aBase = (size_t)bat * 2048 * 2048; bBase = (size_t)bat * 1024 * 2048; }

    // fragment lane mapping (ISA §7.12.2)
    const int arow = lane & 15;
    const int ka   = (lane < 16) ? 0 : 8;
    const int kb   = (lane < 16) ? 0 : 8;

    v8f acc[4][4];
    #pragma unroll
    for (int i = 0; i < 4; i++)
        #pragma unroll
        for (int j = 0; j < 4; j++)
            acc[i][j] = (v8f){0.f, 0.f, 0.f, 0.f, 0.f, 0.f, 0.f, 0.f};

    const int nchunk = K >> 5;

#if USE_TDM
    // wave0 stages the A tile, wave1 the B tile; all operands wave-uniform.
    if (w == 0) tdm_load_2d(A + aBase + (size_t)m0 * K, &As[0][0], 128, (uint32_t)K);
    if (w == 1) tdm_load_2d(B + bBase + (size_t)n0 * K, &Bs[0][0], 256, (uint32_t)K);

    for (int i = 0; i < nchunk; ++i) {
        const int cur = i & 1;
        if (w < 2) {
            if (i + 1 < nchunk) {
                const int kk = (i + 1) << 5;
                if (w == 0)
                    tdm_load_2d(A + aBase + (size_t)m0 * K + kk, &As[cur ^ 1][0], 128, (uint32_t)K);
                else
                    tdm_load_2d(B + bBase + (size_t)n0 * K + kk, &Bs[cur ^ 1][0], 256, (uint32_t)K);
                wait_tensorcnt_le1();          // in-order: tile i resident
            } else {
                wait_tensorcnt_0();
            }
        }
        __syncthreads();                       // publish tile i

        v16bf af[4], bfr[4];
        #pragma unroll
        for (int mt = 0; mt < 4; mt++)
            af[mt] = load_frag_a(&As[cur][(wm * 64 + mt * 16 + arow) * 32 + ka]);
        #pragma unroll
        for (int nt = 0; nt < 4; nt++)
            bfr[nt] = load_frag_b(&Bs[cur][(wn * 64 + nt * 16 + arow) * 32 + kb]);

        #pragma unroll
        for (int mt = 0; mt < 4; mt++)
            #pragma unroll
            for (int nt = 0; nt < 4; nt++)
                acc[mt][nt] = __builtin_amdgcn_wmma_f32_16x16x32_bf16(
                    false, af[mt], false, bfr[nt], (short)0, acc[mt][nt], false, false);

        __syncthreads();   // all reads of buf[cur] done before it is refilled
    }
#elif USE_ASYNC_LDS
    // per-thread copy mapping: A tile 128x32 (2x16B), B tile 256x32 (4x16B)
    const int aRow = tid >> 1, aCol = (tid & 1) * 16;
    auto issue_tile = [&](int buf, int kk) {
        const uint16_t* gA = A + aBase + (size_t)(m0 + aRow) * K + kk + aCol;
        uint16_t* lA = &As[buf][aRow * 32 + aCol];
        async_b128(gA,     lA);
        async_b128(gA + 8, lA + 8);
        const uint16_t* gB = B + bBase + (size_t)(n0 + tid) * K + kk;
        uint16_t* lB = &Bs[buf][tid * 32];
        async_b128(gB,      lB);
        async_b128(gB + 8,  lB + 8);
        async_b128(gB + 16, lB + 16);
        async_b128(gB + 24, lB + 24);
    };

    issue_tile(0, 0);
    for (int i = 0; i < nchunk; ++i) {
        const int cur = i & 1;
        if (i + 1 < nchunk) {
            issue_tile(cur ^ 1, (i + 1) << 5);          // prefetch next tile
            asm volatile("s_wait_asynccnt 0x6" ::: "memory");  // tile i resident
        } else {
            asm volatile("s_wait_asynccnt 0x0" ::: "memory");
        }
        __syncthreads();                                 // publish tile i

        v16bf af[4], bfr[4];
        #pragma unroll
        for (int mt = 0; mt < 4; mt++)
            af[mt] = load_frag_a(&As[cur][(wm * 64 + mt * 16 + arow) * 32 + ka]);
        #pragma unroll
        for (int nt = 0; nt < 4; nt++)
            bfr[nt] = load_frag_b(&Bs[cur][(wn * 64 + nt * 16 + arow) * 32 + kb]);

        #pragma unroll
        for (int mt = 0; mt < 4; mt++)
            #pragma unroll
            for (int nt = 0; nt < 4; nt++)
                acc[mt][nt] = __builtin_amdgcn_wmma_f32_16x16x32_bf16(
                    false, af[mt], false, bfr[nt], (short)0, acc[mt][nt], false, false);

        __syncthreads();
    }
#else
    const int aRow = tid >> 1, aCol = (tid & 1) * 16;
    for (int i = 0; i < nchunk; ++i) {
        const int kk = i << 5;
        const uint4* ga = (const uint4*)(A + aBase + (size_t)(m0 + aRow) * K + kk + aCol);
        const uint4* gb = (const uint4*)(B + bBase + (size_t)(n0 + tid) * K + kk);
        uint4 a0 = ga[0], a1 = ga[1];
        uint4 b0 = gb[0], b1 = gb[1], b2 = gb[2], b3 = gb[3];

        __syncthreads();
        *(uint4*)(&As[0][aRow * 32 + aCol])     = a0;
        *(uint4*)(&As[0][aRow * 32 + aCol + 8]) = a1;
        *(uint4*)(&Bs[0][tid * 32])      = b0;
        *(uint4*)(&Bs[0][tid * 32 + 8])  = b1;
        *(uint4*)(&Bs[0][tid * 32 + 16]) = b2;
        *(uint4*)(&Bs[0][tid * 32 + 24]) = b3;
        __syncthreads();

        v16bf af[4], bfr[4];
        #pragma unroll
        for (int mt = 0; mt < 4; mt++)
            af[mt] = load_frag_a(&As[0][(wm * 64 + mt * 16 + arow) * 32 + ka]);
        #pragma unroll
        for (int nt = 0; nt < 4; nt++)
            bfr[nt] = load_frag_b(&Bs[0][(wn * 64 + nt * 16 + arow) * 32 + kb]);

        #pragma unroll
        for (int mt = 0; mt < 4; mt++)
            #pragma unroll
            for (int nt = 0; nt < 4; nt++)
                acc[mt][nt] = __builtin_amdgcn_wmma_f32_16x16x32_bf16(
                    false, af[mt], false, bfr[nt], (short)0, acc[mt][nt], false, false);
    }
#endif

    // epilogue: C VGPR j -> row (j or j+8), col = lane&15
    const int crow0 = (lane < 16) ? 0 : 8;
    const int cn    = lane & 15;
    #pragma unroll
    for (int mt = 0; mt < 4; mt++)
        #pragma unroll
        for (int nt = 0; nt < 4; nt++)
            #pragma unroll
            for (int j = 0; j < 8; j++) {
                int gm = m0 + wm * 64 + mt * 16 + crow0 + j;
                int gn = n0 + wn * 64 + nt * 16 + cn;
                float v = acc[mt][nt][j];
                if (MODE == 0) {
                    int bb = gm >> 11, s = gm & 2047;    // batch, seq
                    int sel = gn >> 10, o = gn & 1023;   // Q/K/V, feature
                    if (sel == 0) {
                        oQ[((size_t)bb * 2048 + s) * 1024 + o] = f2bf(v + bq[o]);
                    } else if (sel == 1) {
                        oK[((size_t)bb * 2048 + s) * 1024 + o] = f2bf(v + bk[o]);
                    } else {
                        // store V transposed: Vt[b][o][s]
                        oVt[((size_t)bb * 1024 + o) * 2048 + s] = f2bf(v + bv[o]);
                    }
                } else if (MODE == 1) {
                    oF[(size_t)bat * 2048 * 2048 + (size_t)gm * 2048 + gn] = v * 0.03125f;
                } else {
                    oF[(size_t)bat * 2048 * 1024 + (size_t)gm * 1024 + gn] = v;
                }
            }
}

// ---------------------------------------------------------------------------
// Row softmax over 2048 fp32 scores -> bf16 probabilities
// ---------------------------------------------------------------------------
__global__ __launch_bounds__(256)
void softmax_rows(const float* __restrict__ S, uint16_t* __restrict__ P) {
    __shared__ float red[256];
    const int row = blockIdx.x;
    const int t   = threadIdx.x;
    const float* sr = S + (size_t)row * 2048;

    float vals[8];
    float m = -3.402823466e38f;
    #pragma unroll
    for (int j = 0; j < 8; j++) { vals[j] = sr[t + j * 256]; m = fmaxf(m, vals[j]); }
    red[t] = m; __syncthreads();
    for (int s = 128; s > 0; s >>= 1) {
        if (t < s) red[t] = fmaxf(red[t], red[t + s]);
        __syncthreads();
    }
    m = red[0]; __syncthreads();

    float sum = 0.f;
    #pragma unroll
    for (int j = 0; j < 8; j++) { vals[j] = __expf(vals[j] - m); sum += vals[j]; }
    red[t] = sum; __syncthreads();
    for (int s = 128; s > 0; s >>= 1) {
        if (t < s) red[t] += red[t + s];
        __syncthreads();
    }
    float inv = 1.f / red[0];

    uint16_t* pr = P + (size_t)row * 2048;
    #pragma unroll
    for (int j = 0; j < 8; j++) pr[t + j * 256] = f2bf(vals[j] * inv);
}

// ---------------------------------------------------------------------------
extern "C" void kernel_launch(void* const* d_in, const int* in_sizes, int n_in,
                              void* d_out, int out_size, void* d_ws, size_t ws_size,
                              hipStream_t stream) {
    (void)in_sizes; (void)n_in; (void)out_size; (void)ws_size;

    const float* x  = (const float*)d_in[0];
    const float* Wq = (const float*)d_in[1];
    const float* bq = (const float*)d_in[2];
    const float* Wk = (const float*)d_in[3];
    const float* bk = (const float*)d_in[4];
    const float* Wv = (const float*)d_in[5];
    const float* bv = (const float*)d_in[6];
    float* out = (float*)d_out;

    char* ws = (char*)d_ws;
    uint16_t* xbf  = (uint16_t*)(ws);
    uint16_t* wbf  = (uint16_t*)(ws + 16777216);
    uint16_t* qbf  = (uint16_t*)(ws + 23068672);
    uint16_t* kbf  = (uint16_t*)(ws + 39845888);
    uint16_t* vtbf = (uint16_t*)(ws + 56623104);
    float*    sbuf = (float*)   (ws + 73400320);
    uint16_t* pbf  = (uint16_t*)(ws + 140509184);

    // 1) convert inputs to bf16
    cvt_f32_bf16<<<32768, 256, 0, stream>>>(x, xbf, 8388608);
    pack_w<<<12288, 256, 0, stream>>>(Wq, Wk, Wv, wbf);

    // 2) QKV projection: (8192x1024) x (3072x1024)^T
    gemm_nt<0><<<dim3(64, 12, 1), 256, 0, stream>>>(
        xbf, wbf, 1024, bq, bk, bv, qbf, kbf, vtbf, nullptr);

    // 3) scores = Q K^T / 32 : per batch (2048x1024) x (2048x1024)^T
    gemm_nt<1><<<dim3(16, 8, 4), 256, 0, stream>>>(
        qbf, kbf, 1024, nullptr, nullptr, nullptr, nullptr, nullptr, nullptr, sbuf);

    // 4) softmax over key dim
    softmax_rows<<<8192, 256, 0, stream>>>(sbuf, pbf);

    // 5) out = P V : per batch (2048x2048) x (1024x2048)^T  (Vt is h-major)
    gemm_nt<2><<<dim3(16, 4, 4), 256, 0, stream>>>(
        pbf, vtbf, 2048, nullptr, nullptr, nullptr, nullptr, nullptr, nullptr, out);
}
